// TopKDecoder_51556787421290
// MI455X (gfx1250) — matrix operand order repaired and checked by reading, not compiled, not run on hardware
//
#include <hip/hip_runtime.h>
#include <math.h>

// ---------------- problem constants (from reference) ----------------
#define Bsz     32
#define KK      5
#define NN      (Bsz * KK)      // 160 rows, 10 M-tiles
#define HH      512
#define H2      (2 * HH)        // 1024
#define G3      (3 * HH)        // 1536
#define TT      256
#define VV      5000
#define MAXLEN  64
#define EOS_TOK 2
#define PAD_TOK 0
#define NEGV    (-1.0e9f)
#define SCALEF  0.04419417382415922f   // 1/sqrt(512)
#define NTW     4                      // N-tiles per wave in the GEMM

typedef __bf16 bf16_t;
typedef __attribute__((ext_vector_type(16))) __bf16 v16bf;
typedef __attribute__((ext_vector_type(8)))  float  v8f;

// Inverse of the CDNA5 16-bit A-fragment k-pattern:
//   lane = kh*16 + (m&15),  element i = grp*8 + r2,
//   where k = grp*16 + kh*8 + r2  (grp=k>>4, kh=(k>>3)&1, r2=k&7)
__device__ inline size_t a_pack_idx(int r, int c, int ktiles) {
    int mt  = r >> 4, lm = r & 15;
    int kt  = c >> 5, kin = c & 31;
    int grp = kin >> 4, rem = kin & 15;
    int kh  = rem >> 3, r2 = rem & 7;
    int i   = (grp << 3) + r2;
    int lane = (kh << 4) + lm;
    return ((size_t)(mt * ktiles + kt) * 32 + lane) * 16 + i;
}

// ---------------- pack fp32 weights into bf16 B-fragment tiles ----------------
// Wp[((kt*ntnPad + nt)*32 + lane)*16 + i] = W[kt*32 + (i>>3)*16 + (lane>>4)*8 + (i&7)][nt*16 + (lane&15)]
// pad columns (n >= Ncols) are zero-filled, so padded N-tiles contribute nothing.
__global__ void k_packW(const float* __restrict__ W, bf16_t* __restrict__ Wp,
                        int Ncols, int ntnPad, int total) {
    for (int idx = blockIdx.x * blockDim.x + threadIdx.x; idx < total;
         idx += gridDim.x * blockDim.x) {
        int i    = idx & 15;
        int lane = (idx >> 4) & 31;
        int tile = idx >> 9;
        int kt = tile / ntnPad, nt = tile % ntnPad;
        int k = (kt << 5) + ((i >> 3) << 4) + ((lane >> 4) << 3) + (i & 7);
        int n = (nt << 4) + (lane & 15);
        Wp[idx] = (n < Ncols) ? (bf16_t)W[(size_t)k * Ncols + n] : (bf16_t)0.f;
    }
}

__global__ void k_zero(float* __restrict__ d, int n) {
    int i = blockIdx.x * blockDim.x + threadIdx.x;
    if (i < n) d[i] = 0.f;
}

// gather embedding row per token + convert hidden; write both in A-fragment layout
__global__ void k_prep(const int* __restrict__ tok, const float* __restrict__ emb,
                       const float* __restrict__ hcur,
                       bf16_t* __restrict__ xP, bf16_t* __restrict__ hP, int rows) {
    int i = blockIdx.x * blockDim.x + threadIdx.x;
    if (i >= rows * HH) return;
    int r = i / HH, c = i % HH;
    size_t d = a_pack_idx(r, c, HH / 32);
    xP[d] = (bf16_t)emb[(size_t)tok[r] * HH + c];
    hP[d] = (bf16_t)hcur[i];
}

// ---------------- WMMA bf16 GEMM on pre-packed fragments ----------------
// C[M,Ncols] = A[M,Kd] x W[Kd,Ncols]; one wave: 1 M-tile x 4 N-tiles, no k-loop branches.
__global__ void k_gemm(const v16bf* __restrict__ Ap, const v16bf* __restrict__ Wp,
                       float* __restrict__ C, int mtiles, int ntnPad, int ktiles, int Ncols) {
    int wave = (blockIdx.x * blockDim.x + threadIdx.x) >> 5;
    int lane = threadIdx.x & 31;
    int ngroups = ntnPad / NTW;
    if (wave >= mtiles * ngroups) return;
    int mt  = wave / ngroups;
    int nt0 = (wave % ngroups) * NTW;

    const v16bf* ap = Ap + (size_t)mt * ktiles * 32 + lane;
    const v16bf* bp = Wp + (size_t)nt0 * 32 + lane;
    const size_t bstep = (size_t)ntnPad * 32;

    v8f acc0 = {}, acc1 = {}, acc2 = {}, acc3 = {};
#pragma unroll 2
    for (int kt = 0; kt < ktiles; ++kt) {
        v16bf a  = ap[(size_t)kt * 32];       // 32B contiguous per lane
        v16bf b0 = bp[0];
        v16bf b1 = bp[32];
        v16bf b2 = bp[64];
        v16bf b3 = bp[96];
        bp += bstep;
        acc0 = __builtin_amdgcn_wmma_f32_16x16x32_bf16(false, a, false, b0, (short)0, acc0, false, false);
        acc1 = __builtin_amdgcn_wmma_f32_16x16x32_bf16(false, a, false, b1, (short)0, acc1, false, false);
        acc2 = __builtin_amdgcn_wmma_f32_16x16x32_bf16(false, a, false, b2, (short)0, acc2, false, false);
        acc3 = __builtin_amdgcn_wmma_f32_16x16x32_bf16(false, a, false, b3, (short)0, acc3, false, false);
    }

    int kh = lane >> 4, ln = lane & 15;
    int n0 = (nt0 << 4) + ln;
    float* crow = C + (size_t)((mt << 4) + (kh << 3)) * Ncols;
#pragma unroll
    for (int r = 0; r < 8; ++r) {
        size_t off = (size_t)r * Ncols;
        if (n0      < Ncols) crow[off + n0]      = acc0[r];
        if (n0 + 16 < Ncols) crow[off + n0 + 16] = acc1[r];
        if (n0 + 32 < Ncols) crow[off + n0 + 32] = acc2[r];
        if (n0 + 48 < Ncols) crow[off + n0 + 48] = acc3[r];
    }
}

// ---------------- GRU elementwise: h' = (1-z)*n + z*h ----------------
// writes h_next (fp32) and the h-half of the [h,ctx] concat (packed bf16)
__global__ void k_gru(const float* __restrict__ gx, const float* __restrict__ gh,
                      const float* __restrict__ hold, float* __restrict__ hnew,
                      bf16_t* __restrict__ hcP, int rows) {
    int i = blockIdx.x * blockDim.x + threadIdx.x;
    if (i >= rows * HH) return;
    int r = i / HH, c = i % HH;
    const float* gxr = gx + (size_t)r * G3;
    const float* ghr = gh + (size_t)r * G3;
    float z  = 1.f / (1.f + __expf(-(gxr[c] + ghr[c])));
    float rr = 1.f / (1.f + __expf(-(gxr[HH + c] + ghr[HH + c])));
    float nn = tanhf(gxr[2 * HH + c] + rr * ghr[2 * HH + c]);
    float hv = (1.f - z) * nn + z * hold[i];
    hnew[i] = hv;
    hcP[a_pack_idx(r, c, H2 / 32)] = (bf16_t)hv;
}

// ---------------- attention: scores + softmax over T (one block per row) ----------------
__global__ void k_attn(const float* __restrict__ h, const float* __restrict__ enc,
                       float* __restrict__ att, int bpb) {
    int r = blockIdx.x, t = threadIdx.x;   // blockDim.x == TT == 256
    int b = r / bpb;
    const float* hr = h + (size_t)r * HH;
    const float* e  = enc + ((size_t)b * TT + t) * HH;
    float s = 0.f;
    for (int c = 0; c < HH; ++c) s += hr[c] * e[c];
    s *= SCALEF;
    __shared__ float red[TT];
    red[t] = s; __syncthreads();
    for (int o = 128; o > 0; o >>= 1) { if (t < o) red[t] = fmaxf(red[t], red[t + o]); __syncthreads(); }
    float mx = red[0]; __syncthreads();
    float ex = __expf(s - mx);
    red[t] = ex; __syncthreads();
    for (int o = 128; o > 0; o >>= 1) { if (t < o) red[t] += red[t + o]; __syncthreads(); }
    att[(size_t)r * TT + t] = ex / red[0];
}

// ---------------- context: ctx = att @ enc, writes ctx-half of concat (packed bf16) ----
__global__ void k_ctx(const float* __restrict__ att, const float* __restrict__ enc,
                      bf16_t* __restrict__ hcP, int rows, int bpb) {
    int i = blockIdx.x * blockDim.x + threadIdx.x;
    if (i >= rows * HH) return;
    int r = i / HH, c = i % HH;
    int b = r / bpb;
    const float* a = att + (size_t)r * TT;
    const float* e = enc + (size_t)b * TT * HH + c;
    float s = 0.f;
    for (int t = 0; t < TT; ++t) s += a[t] * e[(size_t)t * HH];
    hcP[a_pack_idx(r, HH + c, H2 / 32)] = (bf16_t)s;
}

// ---------------- per-row log-sum-exp; rowbias = cum - lse ----------------
__global__ void k_rowstats(const float* __restrict__ logits, const float* __restrict__ cum,
                           float* __restrict__ rowbias, int useCum) {
    int r = blockIdx.x, t = threadIdx.x;
    const float* L = logits + (size_t)r * VV;
    float mx = -3.4e38f;
    for (int v = t; v < VV; v += 256) mx = fmaxf(mx, L[v]);
    __shared__ float red[256];
    red[t] = mx; __syncthreads();
    for (int o = 128; o > 0; o >>= 1) { if (t < o) red[t] = fmaxf(red[t], red[t + o]); __syncthreads(); }
    mx = red[0]; __syncthreads();
    float s = 0.f;
    for (int v = t; v < VV; v += 256) s += __expf(L[v] - mx);
    red[t] = s; __syncthreads();
    for (int o = 128; o > 0; o >>= 1) { if (t < o) red[t] += red[t + o]; __syncthreads(); }
    if (t == 0) rowbias[r] = (useCum ? cum[r] : 0.f) - (__logf(red[0]) + mx);
}

__device__ inline void top5_insert(float* s, int* id, float sc, int i) {
    if (sc <= s[4]) return;
    int p = 4;
    while (p > 0 && sc > s[p - 1]) { s[p] = s[p - 1]; id[p] = id[p - 1]; --p; }
    s[p] = sc; id[p] = i;
}

// ---------------- step-0 top-k: per batch over V ----------------
__global__ void k_topk0(const float* __restrict__ logits, const float* __restrict__ rowbias,
                        float* __restrict__ cum, int* __restrict__ tokens,
                        int* __restrict__ fin, int* __restrict__ len, int* __restrict__ lastTok) {
    int b = blockIdx.x, t = threadIdx.x;
    float rb = rowbias[b];
    float ls[5]; int li[5];
    for (int q = 0; q < 5; ++q) { ls[q] = -3.4e38f; li[q] = 0; }
    const float* L = logits + (size_t)b * VV;
    for (int v = t; v < VV; v += 256) top5_insert(ls, li, L[v] + rb, v);
    __shared__ float ss[256 * 5]; __shared__ int si[256 * 5];
    for (int q = 0; q < 5; ++q) { ss[t * 5 + q] = ls[q]; si[t * 5 + q] = li[q]; }
    __syncthreads();
    if (t == 0) {
        float bs[5]; int bi[5];
        for (int q = 0; q < 5; ++q) { bs[q] = -3.4e38f; bi[q] = 0; }
        for (int e = 0; e < 256 * 5; ++e) top5_insert(bs, bi, ss[e], si[e]);
        for (int j = 0; j < KK; ++j) {
            int row = b * KK + j; int v = bi[j];
            cum[row] = bs[j];
            tokens[row * MAXLEN + 0] = v;
            fin[row] = (v == EOS_TOK) ? 1 : 0;
            len[row] = 1;
            lastTok[row] = v;
        }
    }
}

// ---------------- joint top-k over k*V candidates (one block per batch) ----------------
__global__ void k_topk(const float* __restrict__ logits, const float* __restrict__ rowbias,
                       float* __restrict__ cum, const int* __restrict__ finOld,
                       int* __restrict__ beamIds, int* __restrict__ tokIds) {
    int b = blockIdx.x, t = threadIdx.x;
    float rb[KK], cm[KK]; int fn[KK];
    for (int j = 0; j < KK; ++j) {
        rb[j] = rowbias[b * KK + j]; cm[j] = cum[b * KK + j]; fn[j] = finOld[b * KK + j];
    }
    float ls[5]; int li[5];
    for (int q = 0; q < 5; ++q) { ls[q] = -3.4e38f; li[q] = 0; }
    for (int cd = t; cd < KK * VV; cd += 256) {
        int j = cd / VV, v = cd - j * VV;
        float sc = fn[j] ? (cm[j] + ((v == PAD_TOK) ? 0.f : NEGV))
                         : (rb[j] + logits[(size_t)(b * KK + j) * VV + v]);
        top5_insert(ls, li, sc, cd);
    }
    __shared__ float ss[256 * 5]; __shared__ int si[256 * 5];
    for (int q = 0; q < 5; ++q) { ss[t * 5 + q] = ls[q]; si[t * 5 + q] = li[q]; }
    __syncthreads();
    if (t == 0) {
        float bs[5]; int bi[5];
        for (int q = 0; q < 5; ++q) { bs[q] = -3.4e38f; bi[q] = 0; }
        for (int e = 0; e < 256 * 5; ++e) top5_insert(bs, bi, ss[e], si[e]);
        for (int j = 0; j < KK; ++j) {
            int cd = bi[j];
            beamIds[b * KK + j] = cd / VV;
            tokIds[b * KK + j]  = cd % VV;
            cum[b * KK + j]     = bs[j];
        }
    }
}

// ---------------- beam-state gathers ----------------
__global__ void k_gather_meta(const int* __restrict__ beamIds, const int* __restrict__ tokIds,
                              const int* __restrict__ tokOld, int* __restrict__ tokNew,
                              const int* __restrict__ finOld, int* __restrict__ finNew,
                              const int* __restrict__ lenOld, int* __restrict__ lenNew,
                              int* __restrict__ lastTok, int di) {
    int i = blockIdx.x * blockDim.x + threadIdx.x;
    if (i >= NN) return;
    int b = i / KK;
    int src = b * KK + beamIds[i];
    for (int l = 0; l < MAXLEN; ++l) tokNew[i * MAXLEN + l] = tokOld[src * MAXLEN + l];
    int tk = tokIds[i];
    tokNew[i * MAXLEN + di] = tk;
    int pf = finOld[src];
    lenNew[i] = lenOld[src] + (pf ? 0 : 1);
    finNew[i] = pf | (tk == EOS_TOK);
    lastTok[i] = tk;
}

__global__ void k_gather_h(const int* __restrict__ beamIds, const float* __restrict__ hNext,
                           float* __restrict__ hCur) {
    int i = blockIdx.x * blockDim.x + threadIdx.x;
    if (i >= NN * HH) return;
    int r = i / HH, c = i % HH;
    int b = r / KK;
    hCur[i] = hNext[(size_t)(b * KK + beamIds[r]) * HH + c];
}

__global__ void k_inflate_h(const float* __restrict__ hNext, float* __restrict__ hCur) {
    int i = blockIdx.x * blockDim.x + threadIdx.x;
    if (i >= NN * HH) return;
    int r = i / HH, c = i % HH;
    hCur[i] = hNext[(size_t)(r / KK) * HH + c];
}

// ---------------- length penalty + best beam -> d_out ----------------
__global__ void k_finalize(const float* __restrict__ cum, const int* __restrict__ fin,
                           const int* __restrict__ len, const int* __restrict__ tokens,
                           float* __restrict__ out) {
    int b = blockIdx.x * blockDim.x + threadIdx.x;
    if (b >= Bsz) return;
    float best = -3.4e38f; int bj = 0;
    for (int j = 0; j < KK; ++j) {
        int row = b * KK + j;
        float pen = powf((5.0f + (float)len[row]) / 6.0f, 1.2f);
        float sel = fin[row] ? cum[row] / pen : cum[row];
        if (sel > best) { best = sel; bj = j; }
    }
    int row = b * KK + bj;
    for (int l = 0; l < MAXLEN; ++l) out[b * MAXLEN + l] = (float)tokens[row * MAXLEN + l];
    for (int j = 0; j < KK; ++j) out[Bsz * MAXLEN + b * KK + j] = cum[b * KK + j];
}

// ------------------------------------------------------------------
extern "C" void kernel_launch(void* const* d_in, const int* in_sizes, int n_in,
                              void* d_out, int out_size, void* d_ws, size_t ws_size,
                              hipStream_t stream) {
    (void)in_sizes; (void)n_in; (void)out_size; (void)ws_size;
    const int*   input_var = (const int*)  d_in[0];
    const float* enc       = (const float*)d_in[1];
    const float* emb       = (const float*)d_in[2];
    const float* w_ih      = (const float*)d_in[3];
    const float* w_hh      = (const float*)d_in[4];
    const float* w_out     = (const float*)d_in[5];

    const int NTN_G  = G3 / 16;                               // 96 (multiple of NTW)
    const int NTN_V  = ((VV + 15) / 16 + NTW - 1) / NTW * NTW; // 313 -> 316 padded
    const int KT_H   = HH / 32;                               // 16 K-tiles (gates)
    const int KT_H2  = H2 / 32;                               // 32 K-tiles (logits)

    char* p = (char*)d_ws;
    auto alloc = [&](size_t bytes) -> void* {
        void* r = (void*)p; p += (bytes + 255) & ~(size_t)255; return r;
    };
    bf16_t* wihP  = (bf16_t*)alloc((size_t)KT_H  * NTN_G * 512 * 2);
    bf16_t* whhP  = (bf16_t*)alloc((size_t)KT_H  * NTN_G * 512 * 2);
    bf16_t* woutP = (bf16_t*)alloc((size_t)KT_H2 * NTN_V * 512 * 2);
    bf16_t* xP    = (bf16_t*)alloc((size_t)NN * HH * 2);
    bf16_t* hP    = (bf16_t*)alloc((size_t)NN * HH * 2);
    bf16_t* hcP   = (bf16_t*)alloc((size_t)NN * H2 * 2);
    float*  hCur  = (float*) alloc((size_t)NN * HH * 4);
    float*  hNext = (float*) alloc((size_t)NN * HH * 4);
    float*  gx    = (float*) alloc((size_t)NN * G3 * 4);
    float*  gh    = (float*) alloc((size_t)NN * G3 * 4);
    float*  att   = (float*) alloc((size_t)NN * TT * 4);
    float*  logits= (float*) alloc((size_t)NN * VV * 4);
    float*  rowbias=(float*) alloc(NN * 4);
    float*  cum   = (float*) alloc(NN * 4);
    int* tokensA  = (int*)   alloc(NN * MAXLEN * 4);
    int* tokensB  = (int*)   alloc(NN * MAXLEN * 4);
    int* finA     = (int*)   alloc(NN * 4);
    int* finB     = (int*)   alloc(NN * 4);
    int* lenA     = (int*)   alloc(NN * 4);
    int* lenB     = (int*)   alloc(NN * 4);
    int* beamIds  = (int*)   alloc(NN * 4);
    int* tokIds   = (int*)   alloc(NN * 4);
    int* lastTok  = (int*)   alloc(NN * 4);

    // once per launch: pack weights into bf16 WMMA B-fragment tiles (L2-resident after)
    k_packW<<<1024, 256, 0, stream>>>(w_ih,  wihP,  G3, NTN_G, KT_H  * NTN_G * 512);
    k_packW<<<1024, 256, 0, stream>>>(w_hh,  whhP,  G3, NTN_G, KT_H  * NTN_G * 512);
    k_packW<<<4096, 256, 0, stream>>>(w_out, woutP, VV, NTN_V, KT_H2 * NTN_V * 512);
    k_zero<<<(NN * HH + 255) / 256, 256, 0, stream>>>(hCur, NN * HH);

    auto run_step = [&](int rows, const int* tokPtr, int bpb) {
        int rh = rows * HH;
        int mt = rows / 16;
        k_prep<<<(rh + 255) / 256, 256, 0, stream>>>(tokPtr, emb, hCur, xP, hP, rows);
        {   // GRU gate GEMMs: [rows,512] x [512,1536]
            int ng = NTN_G / NTW;
            int blocks = (mt * ng * 32 + 255) / 256;
            k_gemm<<<blocks, 256, 0, stream>>>((const v16bf*)xP, (const v16bf*)wihP,
                                               gx, mt, NTN_G, KT_H, G3);
            k_gemm<<<blocks, 256, 0, stream>>>((const v16bf*)hP, (const v16bf*)whhP,
                                               gh, mt, NTN_G, KT_H, G3);
        }
        k_gru<<<(rh + 255) / 256, 256, 0, stream>>>(gx, gh, hCur, hNext, hcP, rows);
        k_attn<<<rows, TT, 0, stream>>>(hNext, enc, att, bpb);
        k_ctx<<<(rh + 255) / 256, 256, 0, stream>>>(att, enc, hcP, rows, bpb);
        {   // logits GEMM: [rows,1024] x [1024,5000]
            int ng = NTN_V / NTW;
            int blocks = (mt * ng * 32 + 255) / 256;
            k_gemm<<<blocks, 256, 0, stream>>>((const v16bf*)hcP, (const v16bf*)woutP,
                                               logits, mt, NTN_V, KT_H2, VV);
        }
    };

    // ---- step 0: un-inflated batch, h = 0 ----
    run_step(Bsz, input_var, 1);
    k_rowstats<<<Bsz, 256, 0, stream>>>(logits, cum, rowbias, 0);
    k_topk0<<<Bsz, 256, 0, stream>>>(logits, rowbias, cum, tokensA, finA, lenA, lastTok);
    k_inflate_h<<<(NN * HH + 255) / 256, 256, 0, stream>>>(hNext, hCur);

    // ---- steps 1..63 on beam-inflated batch ----
    for (int di = 1; di < MAXLEN; ++di) {
        run_step(NN, lastTok, KK);
        k_rowstats<<<NN, 256, 0, stream>>>(logits, cum, rowbias, 1);
        int parity = di & 1;
        int* tokOld = parity ? tokensA : tokensB;
        int* tokNew = parity ? tokensB : tokensA;
        int* finOld = parity ? finA : finB;
        int* finNew = parity ? finB : finA;
        int* lenOld = parity ? lenA : lenB;
        int* lenNew = parity ? lenB : lenA;
        k_topk<<<Bsz, 256, 0, stream>>>(logits, rowbias, cum, finOld, beamIds, tokIds);
        k_gather_meta<<<1, 256, 0, stream>>>(beamIds, tokIds, tokOld, tokNew,
                                             finOld, finNew, lenOld, lenNew, lastTok, di);
        k_gather_h<<<(NN * HH + 255) / 256, 256, 0, stream>>>(beamIds, hNext, hCur);
    }

    // di=63 has parity 1 -> final state lives in the "B" buffers
    k_finalize<<<1, 64, 0, stream>>>(cum, finB, lenB, tokensB, (float*)d_out);
}
